// MultiHeadedAttention_128849019049
// MI455X (gfx1250) — matrix operand order repaired and checked by reading, not compiled, not run on hardware
//
#include <hip/hip_runtime.h>
#include <hip/hip_bf16.h>

// ---------------- problem constants ----------------
#define BB 2
#define SS 2048
#define HH 1024
#define NHH 16
#define DKK 64
#define MM (BB * SS)      // 4096 rows for all projections

typedef __attribute__((ext_vector_type(16))) _Float16 v16h;
typedef __attribute__((ext_vector_type(8)))  _Float16 v8h;
typedef __attribute__((ext_vector_type(8)))  float    v8f;

static __device__ inline v16h cat8(v8h lo, v8h hi) {
    return __builtin_shufflevector(lo, hi, 0,1,2,3,4,5,6,7,8,9,10,11,12,13,14,15);
}
// 16 contiguous halfs (two 16B LDS loads)
static __device__ inline v16h ld16(const _Float16* p) {
    return cat8(*(const v8h*)p, *(const v8h*)(p + 8));
}
static __device__ inline v8f wmma16(v16h a, v16h b, v8f c) {
    return __builtin_amdgcn_wmma_f32_16x16x32_f16(false, a, false, b, (short)0, c, false, false);
}
// CDNA5 async DMA: global -> LDS, 16B per lane, tracked by ASYNCcnt.
static __device__ inline void async_b128(void* lds, const void* g) {
    unsigned l = (unsigned)(size_t)lds;   // low 32 bits of flat shared addr = LDS offset
    asm volatile("global_load_async_to_lds_b128 %0, %1, off" :: "v"(l), "v"(g) : "memory");
}

// ---------------- f32 -> f16 convert (8 elem/thread) ----------------
__global__ void cvt_f32_f16(const float* __restrict__ src, _Float16* __restrict__ dst, int n8) {
    int i = blockIdx.x * blockDim.x + threadIdx.x;
    if (i < n8) {
        const float4 a = ((const float4*)src)[(size_t)i * 2];
        const float4 b = ((const float4*)src)[(size_t)i * 2 + 1];
        v8h o = { (_Float16)a.x, (_Float16)a.y, (_Float16)a.z, (_Float16)a.w,
                  (_Float16)b.x, (_Float16)b.y, (_Float16)b.z, (_Float16)b.w };
        *(v8h*)(dst + (size_t)i * 8) = o;
    }
}

// ---------------- WMMA GEMM: C[M,N] = A[M,K] * W[N,K]^T + bias ----------------
// Block tile 128x128, 8 waves as 4(M)x2(N), each wave 32x64 (8 WMMA accumulators).
// K staged 64 at a time into double-buffered LDS via async-to-LDS DMA.
// headLayout=1: write f16 into [B,NH,S,DK]; else write f32 into [M,N].
__global__ __launch_bounds__(256) void gemm_wmma_f16(
    const _Float16* __restrict__ A, const _Float16* __restrict__ W,
    const float* __restrict__ bias, _Float16* __restrict__ outH,
    float* __restrict__ outF, int headLayout)
{
    const int K = HH, N = HH;
    __shared__ __align__(16) _Float16 As[2][128][64];
    __shared__ __align__(16) _Float16 Ws[2][128][64];

    const int tid  = threadIdx.x;
    const int lane = tid & 31;
    const int w    = tid >> 5;
    const int wm   = w & 3;           // 0..3  (M)
    const int wn   = w >> 2;          // 0..1  (N)
    const int am   = lane & 15;
    const int aoff = (lane < 16) ? 0 : 16;  // A-frag K sub-offset
    const int boff = (lane < 16) ? 0 : 16;  // B-frag K sub-offset
    const int mOfs = (lane < 16) ? 0 : 8;   // C/D row sub-offset

    const int mBlk = blockIdx.y * 128;
    const int nBlk = blockIdx.x * 128;

    v8f acc[2][4] = {};

    const int ldRow = tid >> 1;            // 0..127
    const int ldCol = (tid & 1) * 32;      // halfs

    // stage one K=64 tile of A and W into LDS buffer `buf` (8 async b128/thread)
    auto stage = [&](int buf, int k0) {
        const _Float16* ga = &A[(size_t)(mBlk + ldRow) * K + k0 + ldCol];
        const _Float16* gw = &W[(size_t)(nBlk + ldRow) * K + k0 + ldCol];
        #pragma unroll
        for (int j = 0; j < 4; ++j) {
            async_b128(&As[buf][ldRow][ldCol + j * 8], ga + j * 8);
            async_b128(&Ws[buf][ldRow][ldCol + j * 8], gw + j * 8);
        }
    };
    auto compute = [&](int buf) {
        #pragma unroll
        for (int s = 0; s < 2; ++s) {                 // two K=32 sub-steps
            const int ks = s * 32;
            v16h af0 = cat8(*(const v8h*)&As[buf][wm * 32 +      am][ks + aoff],
                            *(const v8h*)&As[buf][wm * 32 +      am][ks + 8 + aoff]);
            v16h af1 = cat8(*(const v8h*)&As[buf][wm * 32 + 16 + am][ks + aoff],
                            *(const v8h*)&As[buf][wm * 32 + 16 + am][ks + 8 + aoff]);
            #pragma unroll
            for (int t = 0; t < 4; ++t) {
                v16h bf = ld16(&Ws[buf][wn * 64 + t * 16 + am][ks + boff]);
                acc[0][t] = wmma16(af0, bf, acc[0][t]);
                acc[1][t] = wmma16(af1, bf, acc[1][t]);
            }
        }
    };

    stage(0, 0);
    int it = 0;
    for (; it < K / 64 - 1; ++it) {           // steady state: always stage next
        const int buf = it & 1;
        stage(buf ^ 1, (it + 1) * 64);
        asm volatile("s_wait_asynccnt 8" ::: "memory");   // oldest tile complete
        __syncthreads();
        compute(buf);
        __syncthreads();
    }
    {                                          // peeled final tile
        asm volatile("s_wait_asynccnt 0" ::: "memory");
        __syncthreads();
        compute(it & 1);
    }

    // C/D layout: vgpr r -> row M = r + 8*(lane>=16), col N = lane%16
    #pragma unroll
    for (int mi = 0; mi < 2; ++mi) {
        #pragma unroll
        for (int t = 0; t < 4; ++t) {
            const int gn = nBlk + wn * 64 + t * 16 + am;
            const float bv = bias[gn];
            #pragma unroll
            for (int r = 0; r < 8; ++r) {
                const int gm = mBlk + wm * 32 + mi * 16 + r + mOfs;
                const float val = acc[mi][t][r] + bv;
                if (headLayout) {
                    const int b = gm >> 11, srow = gm & (SS - 1);
                    const int h = gn >> 6,  d    = gn & (DKK - 1);
                    outH[(((size_t)b * NHH + h) * SS + srow) * DKK + d] = (_Float16)val;
                } else {
                    outF[(size_t)gm * N + gn] = val;
                }
            }
        }
    }
}

// ---------------- flash attention (per head), WMMA f16 ----------------
// grid: (S/128, B*NH), 256 threads = 8 waves, each wave owns 16 q rows.
// K staged via double-buffered async DMA, V via double-buffered in-LDS transpose
// with next-block global loads overlapped against current-block compute.
__global__ __launch_bounds__(256) void attn_wmma(
    const _Float16* __restrict__ Q, const _Float16* __restrict__ K,
    const _Float16* __restrict__ V, const int* __restrict__ mask,
    _Float16* __restrict__ ctx)
{
    __shared__ __align__(16) _Float16 Ks[2][64][64];
    __shared__ __align__(16) _Float16 Vt[2][64][72];     // transposed V, padded stride
    __shared__ __align__(16) _Float16 Pbuf[8][16][32];   // per-wave P scratch

    const int tid  = threadIdx.x;
    const int lane = tid & 31;
    const int w    = tid >> 5;
    const int am   = lane & 15;
    const int aoff = (lane < 16) ? 0 : 16;
    const int boff = (lane < 16) ? 0 : 16;
    const int mOfs = (lane < 16) ? 0 : 8;

    const int bh = blockIdx.y;
    const int b  = bh >> 4;
    const size_t headBase = (size_t)bh * SS * DKK;
    const _Float16* Qh = Q + headBase;
    const _Float16* Kh = K + headBase;
    const _Float16* Vh = V + headBase;

    const int qRow0 = blockIdx.x * 128 + w * 16;

    // resident Q fragments (two K-slices of DK=64)
    const _Float16* qrow = Qh + (size_t)(qRow0 + am) * DKK;
    const v16h qf0 = cat8(*(const v8h*)&qrow[0  + aoff], *(const v8h*)&qrow[8 + aoff]);
    const v16h qf1 = cat8(*(const v8h*)&qrow[32 + aoff], *(const v8h*)&qrow[40 + aoff]);

    v8f acc0 = {}, acc1 = {}, acc2 = {}, acc3 = {};
    float rowM[8], rowL[8];   // rowM: group-replicated max; rowL: PER-LANE partial sum
    #pragma unroll
    for (int r = 0; r < 8; r++) { rowM[r] = -1e30f; rowL[r] = 0.0f; }

    const int sRow = tid >> 2;           // 0..63
    const int sCol = (tid & 3) * 16;     // 0,16,32,48

    // ---- prologue: V block 0 into regs, K block 0 via async DMA ----
    v8h v0r, v1r;
    {
        const _Float16* vs = Vh + (size_t)sRow * DKK + sCol;
        v0r = *(const v8h*)vs;  v1r = *(const v8h*)(vs + 8);
        const _Float16* ks = Kh + (size_t)sRow * DKK + sCol;
        async_b128(&Ks[0][sRow][sCol],     ks);
        async_b128(&Ks[0][sRow][sCol + 8], ks + 8);
    }

    const int nKB = SS / 64;
    for (int it = 0; it < nKB; ++it) {
        const int kb  = it * 64;
        const int buf = it & 1;

        // commit previously loaded V registers into Vt[buf] (transpose)
        #pragma unroll
        for (int i = 0; i < 8; i++) {
            Vt[buf][sCol + i][sRow]     = v0r[i];
            Vt[buf][sCol + 8 + i][sRow] = v1r[i];
        }
        if (it + 1 < nKB) {
            // overlap next block's K DMA + V global loads with this block's compute
            const _Float16* ks = Kh + (size_t)(kb + 64 + sRow) * DKK + sCol;
            async_b128(&Ks[buf ^ 1][sRow][sCol],     ks);
            async_b128(&Ks[buf ^ 1][sRow][sCol + 8], ks + 8);
            const _Float16* vs = Vh + (size_t)(kb + 64 + sRow) * DKK + sCol;
            v0r = *(const v8h*)vs;  v1r = *(const v8h*)(vs + 8);
            asm volatile("s_wait_asynccnt 2" ::: "memory");   // current K block done
        } else {
            asm volatile("s_wait_asynccnt 0" ::: "memory");
        }
        __syncthreads();

        for (int cl = 0; cl < 64; cl += 32) {
            // scores: two 16x16 kv tiles, accumulate over DK=64 (2 WMMA each)
            v8f s0 = {}, s1 = {};
            s0 = wmma16(qf0, ld16(&Ks[buf][cl      + am][boff]),      s0);
            s1 = wmma16(qf0, ld16(&Ks[buf][cl + 16 + am][boff]),      s1);
            s0 = wmma16(qf1, ld16(&Ks[buf][cl      + am][32 + boff]), s0);
            s1 = wmma16(qf1, ld16(&Ks[buf][cl + 16 + am][32 + boff]), s1);

            const int kv0 = kb + cl + am;
            const int m0  = mask[(size_t)b * SS + kv0];
            const int m1  = mask[(size_t)b * SS + kv0 + 16];

            #pragma unroll
            for (int r = 0; r < 8; r++) {
                float x0 = m0 ? s0[r] * 0.125f : 1e-9f;   // 1/sqrt(64)
                float x1 = m1 ? s1[r] * 0.125f : 1e-9f;
                float mx = fmaxf(x0, x1);
                mx = fmaxf(mx, __shfl_xor(mx, 1));
                mx = fmaxf(mx, __shfl_xor(mx, 2));
                mx = fmaxf(mx, __shfl_xor(mx, 4));
                mx = fmaxf(mx, __shfl_xor(mx, 8));
                const float nm = fmaxf(rowM[r], mx);
                const float p0 = __expf(x0 - nm);
                const float p1 = __expf(x1 - nm);
                const float sc = __expf(rowM[r] - nm);
                rowL[r] = rowL[r] * sc + (p0 + p1);   // per-lane partial; reduced at end
                rowM[r] = nm;
                acc0[r] *= sc; acc1[r] *= sc; acc2[r] *= sc; acc3[r] *= sc;
                const int prow = r + mOfs;
                Pbuf[w][prow][am]      = (_Float16)p0;
                Pbuf[w][prow][16 + am] = (_Float16)p1;
            }
            __asm__ volatile("" ::: "memory");   // LDS is in-order per wave

            const v16h pf = cat8(*(const v8h*)&Pbuf[w][am][aoff],
                                 *(const v8h*)&Pbuf[w][am][8 + aoff]);
            acc0 = wmma16(pf, ld16(&Vt[buf][ 0 + am][cl + boff]), acc0);
            acc1 = wmma16(pf, ld16(&Vt[buf][16 + am][cl + boff]), acc1);
            acc2 = wmma16(pf, ld16(&Vt[buf][32 + am][cl + boff]), acc2);
            acc3 = wmma16(pf, ld16(&Vt[buf][48 + am][cl + boff]), acc3);
        }
        __syncthreads();
    }

    // write context in [B,S,H] layout (un-transpose of heads)
    const int h = bh & 15;
    #pragma unroll
    for (int r = 0; r < 8; r++) {
        float sum = rowL[r];
        sum += __shfl_xor(sum, 1);
        sum += __shfl_xor(sum, 2);
        sum += __shfl_xor(sum, 4);
        sum += __shfl_xor(sum, 8);
        const float inv = 1.0f / sum;
        const int qm = qRow0 + r + mOfs;
        const size_t base = ((size_t)b * SS + qm) * HH + h * DKK + am;
        ctx[base +  0] = (_Float16)(acc0[r] * inv);
        ctx[base + 16] = (_Float16)(acc1[r] * inv);
        ctx[base + 32] = (_Float16)(acc2[r] * inv);
        ctx[base + 48] = (_Float16)(acc3[r] * inv);
    }
}

// ---------------- host launcher ----------------
extern "C" void kernel_launch(void* const* d_in, const int* in_sizes, int n_in,
                              void* d_out, int out_size, void* d_ws, size_t ws_size,
                              hipStream_t stream) {
    const float* query = (const float*)d_in[0];
    const float* key   = (const float*)d_in[1];
    const float* value = (const float*)d_in[2];
    const int*   amask = (const int*)d_in[3];
    const float* Wq = (const float*)d_in[4];  const float* bq = (const float*)d_in[5];
    const float* Wk = (const float*)d_in[6];  const float* bk = (const float*)d_in[7];
    const float* Wv = (const float*)d_in[8];  const float* bv = (const float*)d_in[9];
    const float* Wo = (const float*)d_in[10]; const float* bo = (const float*)d_in[11];

    const size_t nAct = (size_t)MM * HH;   // 4M elements
    const size_t nW   = (size_t)HH * HH;   // 1M elements

    char* ws = (char*)d_ws;
    _Float16* qH    = (_Float16*)(ws);                       // [B,NH,S,DK]
    _Float16* kH    = (_Float16*)(ws + 8ull  * 1024 * 1024);
    _Float16* vH    = (_Float16*)(ws + 16ull * 1024 * 1024);
    _Float16* xq    = (_Float16*)(ws + 24ull * 1024 * 1024); // [M,H] f16
    _Float16* xk    = (_Float16*)(ws + 32ull * 1024 * 1024);
    _Float16* xv    = (_Float16*)(ws + 40ull * 1024 * 1024);
    _Float16* wq16  = (_Float16*)(ws + 48ull * 1024 * 1024);
    _Float16* wk16  = (_Float16*)(ws + 50ull * 1024 * 1024);
    _Float16* wv16  = (_Float16*)(ws + 52ull * 1024 * 1024);
    _Float16* wo16  = (_Float16*)(ws + 54ull * 1024 * 1024);
    _Float16* ctx16 = (_Float16*)(ws + 56ull * 1024 * 1024); // [B,S,H] f16

    const int thr = 256;
    const int bAct = (int)((nAct / 8 + thr - 1) / thr);
    const int bW   = (int)((nW   / 8 + thr - 1) / thr);
    cvt_f32_f16<<<bAct, thr, 0, stream>>>(query, xq,   (int)(nAct / 8));
    cvt_f32_f16<<<bAct, thr, 0, stream>>>(key,   xk,   (int)(nAct / 8));
    cvt_f32_f16<<<bAct, thr, 0, stream>>>(value, xv,   (int)(nAct / 8));
    cvt_f32_f16<<<bW,   thr, 0, stream>>>(Wq,    wq16, (int)(nW / 8));
    cvt_f32_f16<<<bW,   thr, 0, stream>>>(Wk,    wk16, (int)(nW / 8));
    cvt_f32_f16<<<bW,   thr, 0, stream>>>(Wv,    wv16, (int)(nW / 8));
    cvt_f32_f16<<<bW,   thr, 0, stream>>>(Wo,    wo16, (int)(nW / 8));

    dim3 gGemm(HH / 128, MM / 128);   // 8 x 32
    gemm_wmma_f16<<<gGemm, thr, 0, stream>>>(xq, wq16, bq, qH, nullptr, 1);
    gemm_wmma_f16<<<gGemm, thr, 0, stream>>>(xk, wk16, bk, kH, nullptr, 1);
    gemm_wmma_f16<<<gGemm, thr, 0, stream>>>(xv, wv16, bv, vH, nullptr, 1);

    dim3 gAttn(SS / 128, BB * NHH); // 16 x 32
    attn_wmma<<<gAttn, thr, 0, stream>>>(qH, kH, vH, amask, ctx16);

    gemm_wmma_f16<<<gGemm, thr, 0, stream>>>(ctx16, wo16, bo, nullptr, (float*)d_out, 0);
}
